// SVDpp_73993696576172
// MI455X (gfx1250) — matrix operand order repaired and checked by reading, not compiled, not run on hardware
//
#include <hip/hip_runtime.h>

typedef float    vf4  __attribute__((ext_vector_type(4)));
typedef float    v8f  __attribute__((ext_vector_type(8)));
typedef _Float16 v8h  __attribute__((ext_vector_type(8)));
typedef _Float16 v16h __attribute__((ext_vector_type(16)));

__device__ __forceinline__ int lower_bound_i(const int* __restrict__ a, int n, int v) {
    int lo = 0, hi = n;
    while (lo < hi) {
        int m = (lo + hi) >> 1;
        if (a[m] < v) lo = m + 1; else hi = m;
    }
    return lo;
}

// Pool table rows for segment b: sum(table[idx[e]]) * rsqrt(count), for this
// thread's 4 owned dims (t = dim-group 0..15). 16 threads cooperate per row;
// indices are fetched 16-wide and broadcast via wave32 shuffles.
__device__ __forceinline__ vf4 pooled4(const float* __restrict__ table,
                                       const int* __restrict__ idx,
                                       const int* __restrict__ seg,
                                       int T, int b, int t, int grpbase) {
    const int start = lower_bound_i(seg, T, b);
    const int end   = lower_bound_i(seg, T, b + 1);
    vf4 acc = {0.f, 0.f, 0.f, 0.f};
    for (int e = start; e < end; e += 16) {
        const int me    = e + t;
        const int myidx = (me < end) ? idx[me] : 0;
        if (me + 16 < end) __builtin_prefetch(idx + me + 16, 0, 3); // global_prefetch_b8
        int n = end - e; if (n > 16) n = 16;
        for (int j = 0; j < n; ++j) {
            const int ri = __shfl(myidx, grpbase + j, 32);
            acc += *((const vf4*)(table + ((size_t)ri << 6)) + t);  // global_load_b128
        }
    }
    const int cnt = end - start;
    const float inv = (cnt > 0) ? rsqrtf((float)cnt) : 0.0f;
    return acc * inv;
}

__global__ __launch_bounds__(256) void svdpp_fwd_kernel(
    const int* __restrict__ scientist_ids, const int* __restrict__ paper_ids,
    const int* __restrict__ sp_idx, const int* __restrict__ sp_seg,
    const int* __restrict__ sw_idx, const int* __restrict__ sw_seg,
    const float* __restrict__ s_fac, const float* __restrict__ p_fac,
    const float* __restrict__ s_bias, const float* __restrict__ p_bias,
    const float* __restrict__ imp_f, const float* __restrict__ imp_w,
    float* __restrict__ out, int B, int Tsp, int Tsw)
{
    __shared__ __attribute__((aligned(32))) _Float16 combh[16][64]; // (s+yu+yw) tile, f16
    __shared__ __attribute__((aligned(32))) _Float16 pmh[16][64];   // paper tile, f16
    __shared__ float sbpb[16];                                      // s_bias + p_bias

    const int tid     = threadIdx.x;
    const int r       = tid >> 4;       // row within 16-row tile
    const int t       = tid & 15;       // dim group: owns dims [4t, 4t+3]
    const int lane    = tid & 31;       // wave32 lane
    const int grpbase = lane & 16;      // 16-lane group base within wave
    const int b0      = (int)blockIdx.x * 16;
    const int b       = b0 + r;
    const bool valid  = (b < B);

    vf4 comb = {0.f, 0.f, 0.f, 0.f};
    vf4 p4   = {0.f, 0.f, 0.f, 0.f};
    float bias = 0.f;

    if (valid) {
        const int sid = scientist_ids[b];
        const int pid = paper_ids[b];
        const vf4 s4 = *((const vf4*)(s_fac + ((size_t)sid << 6)) + t);
        p4           = *((const vf4*)(p_fac + ((size_t)pid << 6)) + t);
        const vf4 yu = pooled4(imp_f, sp_idx, sp_seg, Tsp, b, t, grpbase);
        const vf4 yw = pooled4(imp_w, sw_idx, sw_seg, Tsw, b, t, grpbase);
        comb = s4 + yu + yw;
        if (t == 0) bias = s_bias[sid] + p_bias[pid];
    }

    #pragma unroll
    for (int i = 0; i < 4; ++i) {
        combh[r][t * 4 + i] = (_Float16)comb[i];
        pmh[r][t * 4 + i]   = (_Float16)p4[i];
    }
    if (t == 0) sbpb[r] = bias;
    __syncthreads();

    // Wave 0: interaction[m] = diag( comb(16x64) x P^T(64x16) ) via two chained
    // v_wmma_f32_16x16x32_f16 (K = 0..31, 32..63). EXEC is all-ones in wave 0.
    if (tid < 32) {
        const int row = lane & 15;      // A: M row; B: N column (both = lane&15)
        const int hi  = lane >> 4;      // lane-half selects K sub-block
        v8f c = {};
        #pragma unroll
        for (int kb = 0; kb < 64; kb += 32) {
            // A 16x32 f16 layout: halfs[0..15] = K koff..koff+15 (contiguous),
            // koff = 16 for upper lane half.
            const int koff = kb + hi * 16;
            const v8h alo = *(const v8h*)&combh[row][koff];
            const v8h ahi = *(const v8h*)&combh[row][koff + 8];
            const v16h a = __builtin_shufflevector(alo, ahi,
                0, 1, 2, 3, 4, 5, 6, 7, 8, 9, 10, 11, 12, 13, 14, 15);
            // B 32x16 f16 layout: lane holds N=lane&15, 16 contiguous K values,
            // upper lane half holds K+16. B[k][n] = P[n][k].
            const int boff = kb + hi * 16;
            const v16h bv = *(const v16h*)&pmh[row][boff];
            c = __builtin_amdgcn_wmma_f32_16x16x32_f16(
                    /*neg_a=*/false, a, /*neg_b=*/false, bv,
                    /*c_mod=*/(short)0, c, /*reuse_a=*/false, /*reuse_b=*/false);
        }
        // Diagonal of C (f32 16x16 layout): (m,m) for m=0..7 -> lane m, vgpr m;
        // for m=8..15 -> lane m+16 (24..31), vgpr m-8.
        const bool lo8 = (lane < 8);
        const bool hi8 = (lane >= 24);
        if (lo8 || hi8) {
            const int m = lo8 ? lane : (lane - 16);
            const int v = lo8 ? lane : (lane - 24);
            float val = c[0];
            #pragma unroll
            for (int i = 1; i < 8; ++i) if (v == i) val = c[i];
            const int ob = b0 + m;
            if (ob < B) out[ob] = val + sbpb[m] + 3.82f;
        }
    }
}

extern "C" void kernel_launch(void* const* d_in, const int* in_sizes, int n_in,
                              void* d_out, int out_size, void* d_ws, size_t ws_size,
                              hipStream_t stream) {
    const int*   scientist_ids = (const int*)d_in[0];
    const int*   paper_ids     = (const int*)d_in[1];
    const int*   sp_idx        = (const int*)d_in[2];
    const int*   sp_seg        = (const int*)d_in[3];
    const int*   sw_idx        = (const int*)d_in[4];
    const int*   sw_seg        = (const int*)d_in[5];
    const float* s_fac         = (const float*)d_in[6];
    const float* p_fac         = (const float*)d_in[7];
    const float* s_bias        = (const float*)d_in[8];
    const float* p_bias        = (const float*)d_in[9];
    const float* imp_f         = (const float*)d_in[10];
    const float* imp_w         = (const float*)d_in[11];

    const int B   = in_sizes[0];
    const int Tsp = in_sizes[2];
    const int Tsw = in_sizes[4];

    const int blocks = (B + 15) / 16;
    svdpp_fwd_kernel<<<blocks, 256, 0, stream>>>(
        scientist_ids, paper_ids, sp_idx, sp_seg, sw_idx, sw_seg,
        s_fac, p_fac, s_bias, p_bias, imp_f, imp_w,
        (float*)d_out, B, Tsp, Tsw);
}
